// Encoder_87308095193130
// MI455X (gfx1250) — matrix operand order repaired and checked by reading, not compiled, not run on hardware
//
#include <hip/hip_runtime.h>
#include <hip/hip_bf16.h>
#include <stdint.h>

// ---------------------------------------------------------------------------
// Types for CDNA5 WMMA (wave32): v16bf = 16xbf16 (8 VGPRs), v8f = 8xf32 acc.
// ---------------------------------------------------------------------------
typedef __attribute__((ext_vector_type(16))) __bf16 v16bf;
typedef __attribute__((ext_vector_type(8)))  float  v8f;

// fp32 -> bf16 round-to-nearest-even. The final merge uses v_perm_b32 so the
// compiler cannot re-lower the pattern into 16-bit sub-register moves:
// pack = perm(ub, ua, 0x07060302) = {ub[31:16], ua[31:16]}.
static __device__ __forceinline__ unsigned int bfrne(unsigned int u) {
    return u + 0x7FFFu + ((u >> 16) & 1u);
}
static __device__ __forceinline__ unsigned int pack_bf16x2(float a, float b) {
    const unsigned int ua = bfrne(__builtin_bit_cast(unsigned int, a));
    const unsigned int ub = bfrne(__builtin_bit_cast(unsigned int, b));
    return __builtin_amdgcn_perm(ub, ua, 0x07060302u);
}

// Fragment-major LDS indexing: tiles are stored in exactly the WMMA VGPR
// order (CDNA5 ISA 7.12.2), so a fragment load is two ds_load_b128.
// A (16x32 bf16): lane l<16 row M=l holds K pairs {0..7,16..23}; lanes 16-31
// hold {8..15,24..31}. For dword t = kpair index (K/2):
//   half = (t>>2)&1, vgpr = (t&3) + ((t>>3)<<2)
static __device__ __forceinline__ int a_dw(int r, int t) {
    return (((r >> 4) * 32) + (((t >> 2) & 1) << 4) + (r & 15)) * 8
           + (t & 3) + ((t >> 3) << 2);
}
// B (32x16 bf16): lanes 0-15 col N=l hold K pairs {0..7,16..23}; lanes 16-31
// hold {8..15,24..31}.  For dword t = K/2:
//   half = (t>>2)&1, vgpr = (t&3) + ((t>>3)<<2)
static __device__ __forceinline__ int b_dw(int c, int t) {
    return (((c >> 4) * 32) + (((t >> 2) & 1) << 4) + (c & 15)) * 8
           + (t & 3) + ((t >> 3) << 2);
}

// ---------------------------------------------------------------------------
// Tiled GEMM: C[M,N] = A[M,K] * B[K,N] (+ epilogue), bf16 WMMA, f32 acc.
//   BT=false: B row-major [K,N];  BT=true: B row-major [N,K] -> A*B^T
//   FUSE: 0 none, 1 ReLU, 2 residual add (R, same layout as C)
// Batched over blockIdx.z (z = zb*Hdiv + zh) with per-operand offsets.
// 256 threads = 8 wave32s, register-prefetch + double-buffered LDS,
// one barrier per K-step, L2 prefetch two K-steps ahead (branchless clamp).
// All leading dims / intra-tile offsets are 32-bit (max index < 2^24).
// Non-BT B staging: each active thread owns a 4x4 (k x n) micro-tile so all
// LDS writes are aligned uint2 stores (no 16-bit scatters anywhere).
// ---------------------------------------------------------------------------
template<int BM, int BN, int WTM, int WTN, bool BT, int FUSE>
__global__ __launch_bounds__(256)
void gemm_wmma_kernel(const float* __restrict__ Ag, const float* __restrict__ Bg,
                      float* __restrict__ Cg, const float* __restrict__ Rg,
                      int M, int N, int K,
                      int lda, int ldb, int ldc,
                      int Hdiv,
                      long sAb, long sAh, long sBb, long sBh, long sCb, long sCh)
{
    constexpr int BK = 32;
    constexpr int WAVES_M = BM / WTM;
    constexpr int WAVES_N = BN / WTN;
    static_assert(WAVES_M * WAVES_N == 8, "8 wave32s per block");
    constexpr int IM = WTM / 16;
    constexpr int IN = WTN / 16;
    constexpr int ADW = BM * 16;                 // dwords per A tile buffer
    constexpr int BDW = BN * 16;                 // dwords per B tile buffer
    constexpr int PFA = BM * BK / 4 / 256;       // float4 prefetch regs (A)
    constexpr int PFB = 4;                       // float4 prefetch regs (B)
    constexpr int G   = BN / 4;                  // column-quad groups (non-BT)

    __shared__ __align__(16) unsigned int As[2][ADW];
    __shared__ __align__(16) unsigned int Bs[2][BDW];

    const int tid   = threadIdx.x;
    const int lane  = tid & 31;
    const int wid   = tid >> 5;
    const int waveM = wid % WAVES_M;
    const int waveN = wid / WAVES_M;
    const int lhalf = lane >> 4;
    const int lm    = lane & 15;

    const int z  = blockIdx.z;
    const int zb = z / Hdiv;
    const int zh = z % Hdiv;
    const float* A  = Ag + (long)zb * sAb + (long)zh * sAh;
    const float* Bp = Bg + (long)zb * sBb + (long)zh * sBh;
    float*       C  = Cg + (long)zb * sCb + (long)zh * sCh;
    const float* R  = (FUSE == 2) ? (Rg + (long)zb * sCb + (long)zh * sCh) : nullptr;

    const int row0 = blockIdx.y * BM;
    const int col0 = blockIdx.x * BN;

    v8f acc[IM][IN];
    #pragma unroll
    for (int i = 0; i < IM; ++i)
        #pragma unroll
        for (int j = 0; j < IN; ++j)
            #pragma unroll
            for (int e = 0; e < 8; ++e) acc[i][j][e] = 0.0f;

    float4 pa[PFA], pb[PFB];

    // ---- global -> registers (32-bit offsets) ----
    auto load_global = [&](int k0) {
        #pragma unroll
        for (int i = 0; i < PFA; ++i) {
            const int idx = tid + i * 256;
            const int r  = idx >> 3;             // BK/4 == 8 float4 per row
            const int c4 = (idx & 7) << 2;
            pa[i] = *reinterpret_cast<const float4*>(
                A + ((row0 + r) * lda + (k0 + c4)));
        }
        if (BT) {
            #pragma unroll
            for (int i = 0; i < PFB; ++i) {
                const int idx = tid + i * 256;
                const int r  = idx >> 3;         // n-row
                const int c4 = (idx & 7) << 2;   // k offset
                pb[i] = *reinterpret_cast<const float4*>(
                    Bp + ((col0 + r) * ldb + (k0 + c4)));
            }
        } else {
            // 4x4 micro-tile per active thread: 4 k-rows, one column-quad
            const int cg = tid % G;
            const int kq = tid / G;
            if (kq < 8) {
                #pragma unroll
                for (int j = 0; j < 4; ++j)
                    pb[j] = *reinterpret_cast<const float4*>(
                        Bp + ((k0 + kq * 4 + j) * ldb + (col0 + cg * 4)));
            }
        }
    };

    // ---- registers -> LDS (fp32 -> bf16, fragment-major swizzle) ----
    auto store_lds = [&](int buf) {
        #pragma unroll
        for (int i = 0; i < PFA; ++i) {
            const int idx = tid + i * 256;
            const int r  = idx >> 3;
            const int t0 = ((idx & 7) << 2) >> 1;        // even kpair
            uint2 w;
            w.x = pack_bf16x2(pa[i].x, pa[i].y);
            w.y = pack_bf16x2(pa[i].z, pa[i].w);
            *reinterpret_cast<uint2*>(&As[buf][a_dw(r, t0)]) = w;
        }
        if (BT) {
            #pragma unroll
            for (int i = 0; i < PFB; ++i) {
                const int idx = tid + i * 256;
                const int r  = idx >> 3;                 // n-row (column of B)
                const int t0 = ((idx & 7) << 2) >> 1;    // even kpair
                uint2 w;
                w.x = pack_bf16x2(pb[i].x, pb[i].y);
                w.y = pack_bf16x2(pb[i].z, pb[i].w);
                *reinterpret_cast<uint2*>(&Bs[buf][b_dw(r, t0)]) = w;
            }
        } else {
            const int cg = tid % G;
            const int kq = tid / G;
            if (kq < 8) {
                const int t0 = kq * 2;                   // even kpair
                const float* f0 = reinterpret_cast<const float*>(&pb[0]);
                const float* f1 = reinterpret_cast<const float*>(&pb[1]);
                const float* f2 = reinterpret_cast<const float*>(&pb[2]);
                const float* f3 = reinterpret_cast<const float*>(&pb[3]);
                #pragma unroll
                for (int c = 0; c < 4; ++c) {
                    uint2 w;
                    w.x = pack_bf16x2(f0[c], f1[c]);     // k = 4kq,   4kq+1
                    w.y = pack_bf16x2(f2[c], f3[c]);     // k = 4kq+2, 4kq+3
                    *reinterpret_cast<uint2*>(
                        &Bs[buf][b_dw(cg * 4 + c, t0)]) = w;
                }
            }
        }
    };

    union Frag { uint4 q[2]; v16bf v; };

    const int nK = K / BK;
    load_global(0);
    store_lds(0);
    __syncthreads();

    for (int kb = 0; kb < nK; ++kb) {
        const int  cur  = kb & 1;
        const bool more = (kb + 1) < nK;
        if (more) load_global((kb + 1) * BK);

        // L2 prefetch two K-steps ahead: one cacheline (32 floats) per thread.
        // Branchless clamp: re-prefetching the last tile is harmless.
        {
            int kn = kb + 2; if (kn >= nK) kn = nK - 1;
            const int kp = kn * BK;
            if (tid < 128) {
                __builtin_prefetch(A + ((row0 + tid) * lda + kp), 0, 3);
            } else {
                const int t = tid - 128;
                if (BT) {
                    if (t < BN)
                        __builtin_prefetch(Bp + ((col0 + t) * ldb + kp), 0, 3);
                } else {
                    constexpr int LPR = BN / 32;     // cachelines per k-row
                    const int kr = t / LPR;
                    const int c  = (t % LPR) * 32;
                    if (kr < BK)
                        __builtin_prefetch(Bp + ((kp + kr) * ldb + (col0 + c)), 0, 3);
                }
            }
        }

        Frag afrag[IM], bfrag[IN];
        #pragma unroll
        for (int im = 0; im < IM; ++im) {
            const uint4* p = reinterpret_cast<const uint4*>(
                &As[cur][((waveM * IM + im) * 32 + lane) * 8]);
            afrag[im].q[0] = p[0];
            afrag[im].q[1] = p[1];
        }
        #pragma unroll
        for (int in = 0; in < IN; ++in) {
            const uint4* p = reinterpret_cast<const uint4*>(
                &Bs[cur][((waveN * IN + in) * 32 + lane) * 8]);
            bfrag[in].q[0] = p[0];
            bfrag[in].q[1] = p[1];
        }

        #pragma unroll
        for (int im = 0; im < IM; ++im)
            #pragma unroll
            for (int in = 0; in < IN; ++in)
                acc[im][in] = __builtin_amdgcn_wmma_f32_16x16x32_bf16(
                    false, afrag[im].v, false, bfrag[in].v,
                    (short)0, acc[im][in], false, false);

        if (more) store_lds((kb + 1) & 1);
        __syncthreads();
    }

    // ---- epilogue: C/D f32 layout -> VGPR e: M = e + 8*lhalf, N = lm ----
    #pragma unroll
    for (int im = 0; im < IM; ++im) {
        #pragma unroll
        for (int in = 0; in < IN; ++in) {
            #pragma unroll
            for (int e = 0; e < 8; ++e) {
                const int row = row0 + waveM * WTM + im * 16 + e + 8 * lhalf;
                const int col = col0 + waveN * WTN + in * 16 + lm;
                const int idx = row * ldc + col;     // < 2^24, 32-bit safe
                float val = acc[im][in][e];
                if constexpr (FUSE == 1) val = fmaxf(val, 0.0f);
                if constexpr (FUSE == 2) val += R[idx];
                C[idx] = val;
            }
        }
    }
}

// ---------------------------------------------------------------------------
// x[row,:] = emb[enc[row],:] + pe[row % L,:]
// ---------------------------------------------------------------------------
__global__ __launch_bounds__(256)
void embed_kernel(const int* __restrict__ enc, const float* __restrict__ emb,
                  const float* __restrict__ pe, float* __restrict__ x)
{
    const int row = blockIdx.x;
    const int tok = enc[row];
    const float* e = emb + (long)tok * 1024;
    const float* p = pe + (long)(row & 1023) * 1024;
    float* o = x + (long)row * 1024;
    for (int c = threadIdx.x; c < 1024; c += 256) o[c] = e[c] + p[c];
}

// ---------------------------------------------------------------------------
// In-place masked softmax over one attention row of length L=1024.
// s = raw*0.125; pad (key token == 0) -> -1e9; softmax.
// grid = (L, B*H), block = 256.
// ---------------------------------------------------------------------------
__global__ __launch_bounds__(256)
void softmax_mask_kernel(float* __restrict__ attn, const int* __restrict__ enc)
{
    const int by = blockIdx.y;            // b*16 + h
    const int b  = by >> 4;
    const int qi = blockIdx.x;
    float* row = attn + ((long)by * 1024 + qi) * 1024;
    const int* encb = enc + b * 1024;

    __shared__ float red[256];
    const int t = threadIdx.x;
    float vals[4];
    float mx = -3.402823e38f;
    #pragma unroll
    for (int j = 0; j < 4; ++j) {
        const int n = t + j * 256;
        float s = row[n] * 0.125f;
        if (encb[n] == 0) s = -1e9f;
        vals[j] = s;
        mx = fmaxf(mx, s);
    }
    red[t] = mx; __syncthreads();
    for (int s = 128; s > 0; s >>= 1) {
        if (t < s) red[t] = fmaxf(red[t], red[t + s]);
        __syncthreads();
    }
    mx = red[0]; __syncthreads();

    float sum = 0.0f;
    #pragma unroll
    for (int j = 0; j < 4; ++j) { vals[j] = __expf(vals[j] - mx); sum += vals[j]; }
    red[t] = sum; __syncthreads();
    for (int s = 128; s > 0; s >>= 1) {
        if (t < s) red[t] += red[t + s];
        __syncthreads();
    }
    const float inv = 1.0f / red[0];
    #pragma unroll
    for (int j = 0; j < 4; ++j) row[t + j * 256] = vals[j] * inv;
}

// ---------------------------------------------------------------------------
// LayerNorm (no affine), row length 1024, eps 1e-5. grid = rows, block = 256.
// ---------------------------------------------------------------------------
__global__ __launch_bounds__(256)
void layernorm_kernel(const float* __restrict__ in, float* __restrict__ out)
{
    const int row = blockIdx.x;
    const float* p = in + (long)row * 1024;
    float* o = out + (long)row * 1024;

    __shared__ float r1[256], r2[256];
    const int t = threadIdx.x;
    float v[4], s = 0.0f, s2 = 0.0f;
    #pragma unroll
    for (int j = 0; j < 4; ++j) {
        v[j] = p[t + j * 256];
        s += v[j]; s2 += v[j] * v[j];
    }
    r1[t] = s; r2[t] = s2; __syncthreads();
    for (int st = 128; st > 0; st >>= 1) {
        if (t < st) { r1[t] += r1[t + st]; r2[t] += r2[t + st]; }
        __syncthreads();
    }
    const float mu  = r1[0] * (1.0f / 1024.0f);
    const float var = r2[0] * (1.0f / 1024.0f) - mu * mu;
    const float rs  = rsqrtf(var + 1e-5f);
    #pragma unroll
    for (int j = 0; j < 4; ++j) o[t + j * 256] = (v[j] - mu) * rs;
}

// ---------------------------------------------------------------------------
// Orchestration
// ---------------------------------------------------------------------------
extern "C" void kernel_launch(void* const* d_in, const int* in_sizes, int n_in,
                              void* d_out, int out_size, void* d_ws, size_t ws_size,
                              hipStream_t stream)
{
    (void)in_sizes; (void)n_in; (void)out_size; (void)ws_size;

    const int*   enc = (const int*)  d_in[0];
    const float* emb = (const float*)d_in[1];
    const float* pe  = (const float*)d_in[2];
    const float* WQ  = (const float*)d_in[3];
    const float* WK  = (const float*)d_in[4];
    const float* WV  = (const float*)d_in[5];
    const float* WO  = (const float*)d_in[6];
    const float* W1  = (const float*)d_in[7];
    const float* W2  = (const float*)d_in[8];
    float* out = (float*)d_out;

    constexpr int  Bb = 4, L = 1024, D = 1024, H = 16, DK = 64, DFF = 4096, NL = 2;
    constexpr long M   = (long)Bb * L;       // 4096 rows
    constexpr long XSZ = M * D;              // 4M floats

    // workspace: [x 16MB][y 16MB][q|k|v|ctx 64MB, reused as FFN hidden] = 96MB
    float* x   = (float*)d_ws;
    float* y   = x + XSZ;
    float* q   = y + XSZ;
    float* k   = q + XSZ;
    float* v   = k + XSZ;
    float* ctx = v + XSZ;
    float* hb  = q;                          // M*DFF floats == 4*XSZ exactly

    float* attn_base = out + XSZ;            // attns region of d_out

    embed_kernel<<<dim3((unsigned)M), 256, 0, stream>>>(enc, emb, pe, x);

    for (int i = 0; i < NL; ++i) {
        const float* wq = WQ + (long)i * D * (H * DK);
        const float* wk = WK + (long)i * D * (H * DK);
        const float* wv = WV + (long)i * D * (H * DK);
        const float* wo = WO + (long)i * (H * DK) * D;
        const float* w1 = W1 + (long)i * D * DFF;
        const float* w2 = W2 + (long)i * DFF * D;
        float* attn_l = attn_base + (long)i * Bb * H * (long)L * L;

        // Q, K, V projections: [4096,1024] x [1024,1024]
        gemm_wmma_kernel<128,128,32,64,false,0>
            <<<dim3(D/128, M/128, 1), 256, 0, stream>>>(
            x, wq, q, nullptr, (int)M, D, D, D, D, D, 1, 0,0,0,0,0,0);
        gemm_wmma_kernel<128,128,32,64,false,0>
            <<<dim3(D/128, M/128, 1), 256, 0, stream>>>(
            x, wk, k, nullptr, (int)M, D, D, D, D, D, 1, 0,0,0,0,0,0);
        gemm_wmma_kernel<128,128,32,64,false,0>
            <<<dim3(D/128, M/128, 1), 256, 0, stream>>>(
            x, wv, v, nullptr, (int)M, D, D, D, D, D, 1, 0,0,0,0,0,0);

        // scores = Q @ K^T per (b,h), written straight into d_out attns slice
        gemm_wmma_kernel<128,128,32,64,true,0>
            <<<dim3(L/128, L/128, Bb*H), 256, 0, stream>>>(
            q, k, attn_l, nullptr, L, L, DK, D, D, L, H,
            (long)L*D, (long)DK, (long)L*D, (long)DK,
            (long)H*L*L, (long)L*L);

        softmax_mask_kernel<<<dim3(L, Bb*H), 256, 0, stream>>>(attn_l, enc);

        // ctx = attn @ V per (b,h): [1024,1024] x [1024,64]
        gemm_wmma_kernel<128,64,32,32,false,0>
            <<<dim3(1, L/128, Bb*H), 256, 0, stream>>>(
            attn_l, v, ctx, nullptr, L, DK, L, L, D, D, H,
            (long)H*L*L, (long)L*L, (long)L*D, (long)DK,
            (long)L*D, (long)DK);

        // y = ctx @ WO + x ; x = LN(y)
        gemm_wmma_kernel<128,128,32,64,false,2>
            <<<dim3(D/128, M/128, 1), 256, 0, stream>>>(
            ctx, wo, y, x, (int)M, D, D, D, D, D, 1, 0,0,0,0,0,0);
        layernorm_kernel<<<dim3((unsigned)M), 256, 0, stream>>>(y, x);

        // hb = relu(x @ W1) ; y = hb @ W2 + x ; x = LN(y)
        gemm_wmma_kernel<128,128,32,64,false,1>
            <<<dim3(DFF/128, M/128, 1), 256, 0, stream>>>(
            x, w1, hb, nullptr, (int)M, DFF, D, D, DFF, DFF, 1, 0,0,0,0,0,0);
        gemm_wmma_kernel<128,128,32,64,false,2>
            <<<dim3(D/128, M/128, 1), 256, 0, stream>>>(
            hb, w2, y, x, (int)M, D, DFF, DFF, D, D, 1, 0,0,0,0,0,0);
        layernorm_kernel<<<dim3((unsigned)M), 256, 0, stream>>>(
            y, (i == NL - 1) ? out : x);
    }
}
